// Net1_12532714570032
// MI455X (gfx1250) — matrix operand order repaired and checked by reading, not compiled, not run on hardware
//
#include <hip/hip_runtime.h>
#include <hip/hip_bf16.h>

typedef __attribute__((ext_vector_type(16))) __bf16 v16bf;
typedef __attribute__((ext_vector_type(8)))  float  v8f;

#define IN_F   1433
#define OUT_F  16
#define K_STEP 32

// ---------------------------------------------------------------------------
// Kernel 1: h[n, 0:16] = features[n, :] @ W^T  via v_wmma_f32_16x16x32_bf16
// One wave (32 lanes) computes a 16x16 output tile (16 nodes x 16 out feats).
// ---------------------------------------------------------------------------
__global__ void gcn_gemm_wmma(const float* __restrict__ feat,
                              const float* __restrict__ W,
                              float* __restrict__ h,
                              int n_nodes) {
    const int wave = (int)((blockIdx.x * (unsigned)blockDim.x + threadIdx.x) >> 5);
    const int lane = threadIdx.x & 31;
    const int row_base = wave * 16;
    if (row_base >= n_nodes) return;          // wave-uniform: EXEC stays all-1s

    const int half = lane >> 4;               // 0: lanes 0-15, 1: lanes 16-31
    const int l16  = lane & 15;

    // A fragment addressing (16-bit A 16x32 layout):
    //   lane holds row M = l16.
    //   half==0: elems 0..7 -> K = 0..7,   elems 8..15 -> K = 16..23
    //   half==1: elems 0..7 -> K = 8..15,  elems 8..15 -> K = 24..31
    const float* arow   = feat + (size_t)(row_base + l16) * IN_F;
    const int    a_koff = half * 8;

    // B fragment addressing (16-bit B 32x16 layout, B = W^T):
    //   lane holds column N = l16 (i.e. W row l16), same K order as A
    const float* brow   = W + (size_t)l16 * IN_F;
    const int    b_koff = half * 8;

    v8f acc = {};

    int k0 = 0;
    for (; k0 + K_STEP <= IN_F; k0 += K_STEP) {
        v16bf a, b;
        const float* ap = arow + k0 + a_koff;
#pragma unroll
        for (int i = 0; i < 8; ++i)
            a[i] = (__bf16)__builtin_nontemporal_load(ap + i);
#pragma unroll
        for (int i = 0; i < 8; ++i)
            a[8 + i] = (__bf16)__builtin_nontemporal_load(ap + 16 + i);

        const float* bp = brow + k0 + b_koff;
#pragma unroll
        for (int i = 0; i < 8; ++i)
            b[i] = (__bf16)bp[i];             // W is tiny -> regular (cached) loads
#pragma unroll
        for (int i = 0; i < 8; ++i)
            b[8 + i] = (__bf16)bp[16 + i];

        acc = __builtin_amdgcn_wmma_f32_16x16x32_bf16(
            false, a, false, b, (short)0, acc, false, false);
    }

    // Guarded tail (1433 = 44*32 + 25). Clamp addresses (always in-bounds, so
    // loads are unconditional / speculatable) and zero the value with a VALU
    // select -- no EXEC-mask branching.
    {
        v16bf a, b;
#pragma unroll
        for (int i = 0; i < 8; ++i) {
            int   ka  = k0 + a_koff + i;
            float va  = arow[(ka < IN_F) ? ka : (IN_F - 1)];
            a[i]      = (__bf16)((ka < IN_F) ? va : 0.0f);
            int   kb  = k0 + a_koff + 16 + i;
            float vb  = arow[(kb < IN_F) ? kb : (IN_F - 1)];
            a[8 + i]  = (__bf16)((kb < IN_F) ? vb : 0.0f);
        }
#pragma unroll
        for (int i = 0; i < 16; ++i) {
            int   kk = k0 + b_koff + i + (i >= 8 ? 8 : 0);
            float v  = brow[(kk < IN_F) ? kk : (IN_F - 1)];
            b[i]     = (__bf16)((kk < IN_F) ? v : 0.0f);
        }
        acc = __builtin_amdgcn_wmma_f32_16x16x32_bf16(
            false, a, false, b, (short)0, acc, false, false);
    }

    // C layout: VGPR r -> M = r + 8*half, N = l16.
    float* out = h + (size_t)row_base * OUT_F + l16;
#pragma unroll
    for (int r = 0; r < 8; ++r)
        out[(size_t)(r + 8 * half) * OUT_F] = acc[r];
}

// ---------------------------------------------------------------------------
// Kernel 2: zero the accumulator output (d_out is poisoned by the harness).
// ---------------------------------------------------------------------------
__global__ void gcn_zero(float* __restrict__ out, int n) {
    int i = blockIdx.x * blockDim.x + threadIdx.x;
    if (i < n) out[i] = 0.0f;
}

// ---------------------------------------------------------------------------
// Kernel 3: segment-sum via f32 atomics. 16 lanes per edge (feat = tid & 15),
// so both the h[src] gather and the atomic destination are 64B-contiguous.
// h (6.4 MB) and d_out live in the 192 MB L2 -> atomics resolve in L2.
// ---------------------------------------------------------------------------
__global__ void gcn_scatter(const float* __restrict__ h,
                            const int* __restrict__ src,
                            const int* __restrict__ dst,
                            float* __restrict__ out,
                            long long n_work) {
    long long t = blockIdx.x * (long long)blockDim.x + threadIdx.x;
    if (t >= n_work) return;
    int e = (int)(t >> 4);
    int j = (int)(t & 15);
    int s = src[e];
    int d = dst[e];
    float v = h[(size_t)s * OUT_F + j];
    atomicAdd(out + (size_t)d * OUT_F + j, v);
}

// ---------------------------------------------------------------------------
// Kernel 4: ReLU in place (separate launch = ordering after all atomics).
// ---------------------------------------------------------------------------
__global__ void gcn_relu(float* __restrict__ out, int n) {
    int i = blockIdx.x * blockDim.x + threadIdx.x;
    if (i < n) out[i] = fmaxf(out[i], 0.0f);
}

// ---------------------------------------------------------------------------
extern "C" void kernel_launch(void* const* d_in, const int* in_sizes, int n_in,
                              void* d_out, int out_size, void* d_ws, size_t ws_size,
                              hipStream_t stream) {
    const float* feat = (const float*)d_in[0];   // [N, 1433] f32
    const float* W    = (const float*)d_in[1];   // [16, 1433] f32
    const int*   src  = (const int*)d_in[2];     // [E] i32
    const int*   dst  = (const int*)d_in[3];     // [E] i32
    float*       out  = (float*)d_out;           // [N, 16] f32
    float*       h    = (float*)d_ws;            // [N, 16] f32 scratch (6.4 MB)

    const int n_nodes = in_sizes[0] / IN_F;
    const int n_edges = in_sizes[2];

    // K1: WMMA GEMM, one wave per 16-row tile, 8 waves (256 threads) per block.
    {
        int tiles  = (n_nodes + 15) / 16;
        int blocks = (tiles + 7) / 8;
        gcn_gemm_wmma<<<blocks, 256, 0, stream>>>(feat, W, h, n_nodes);
    }
    // K2: zero output accumulator.
    {
        int n = n_nodes * OUT_F;
        gcn_zero<<<(n + 255) / 256, 256, 0, stream>>>(out, n);
    }
    // K3: edge scatter-add, 16 lanes per edge.
    {
        long long work   = (long long)n_edges * OUT_F;
        long long blocks = (work + 255) / 256;
        gcn_scatter<<<(unsigned)blocks, 256, 0, stream>>>(h, src, dst, out, work);
    }
    // K4: ReLU.
    {
        int n = n_nodes * OUT_F;
        gcn_relu<<<(n + 255) / 256, 256, 0, stream>>>(out, n);
    }
}
